// HeteroRGCN_37821482008831
// MI455X (gfx1250) — matrix operand order repaired and checked by reading, not compiled, not run on hardware
//
#include <hip/hip_runtime.h>
#include <hip/hip_bf16.h>

// ---------------------------------------------------------------------------
// HeteroRGCN forward for MI455X (gfx1250, wave32, WMMA).
//   h = per-type proj (34x GEMM 2048x2048x256)          -> bf16 WMMA, f32 acc
//   per layer: scatter A[dst,b,:] += coeff_b * h[src]   -> f32 global atomics
//              out = [h|A] @ [root;basis] + bias        -> one K=1280 WMMA GEMM
//              LayerNorm + ReLU fused in GEMM epilogue
// GEMMs: 2-stage pipeline (register-staged global loads for k+1 overlap the
// 16 WMMAs of step k), double-buffered LDS (1 barrier/step), B fragments
// software-pipelined under WMMA. B staging packs (k,k+1) bf16 pairs so the
// transposed tile is written with ds_store_b32 (16/thread) not b16 (32/thread).
// ---------------------------------------------------------------------------

#define LDT 40  // LDS row stride in bf16 elems (80B: 16B-aligned, conflict-light)

typedef __attribute__((ext_vector_type(8)))  __bf16 v8bf;
typedef __attribute__((ext_vector_type(16))) __bf16 v16bf;
typedef __attribute__((ext_vector_type(8)))  float  v8f;

static constexpr int  N_ = 69632, T_ = 34, D_ = 2048, H_ = 256, B_ = 4, R_ = 40;
static constexpr int  NPER = 2048;              // N_/T_
static constexpr long E_ = 786432;

__device__ __forceinline__ unsigned short f2bf(float f) {
  unsigned u = __float_as_uint(f);
  unsigned r = u + 0x7FFFu + ((u >> 16) & 1u);  // round-to-nearest-even
  return (unsigned short)(r >> 16);
}
__device__ __forceinline__ unsigned pack2bf(float lo, float hi) {
  return (unsigned)f2bf(lo) | ((unsigned)f2bf(hi) << 16);
}

// ---- register-staged tile loads (128x32 A, 32x256 B), f32 from global ------
__device__ __forceinline__ void glob_loadA(const float* __restrict__ base, int ldg,
                                           int tid, float4 a[4]) {
  const int m = tid >> 3, kq = (tid & 7) << 2;
  #pragma unroll
  for (int i = 0; i < 4; ++i)
    a[i] = *(const float4*)(base + (size_t)(m + (i << 5)) * ldg + kq);
}
// thread owns row pair (2p, 2p+1), cols n4..n4+3;  p = (tid>>6) + 4i
__device__ __forceinline__ void glob_loadB(const float* __restrict__ base, int ldg,
                                           int tid, float4 b[8]) {
  const int n4 = (tid & 63) << 2, pb = tid >> 6;
  #pragma unroll
  for (int i = 0; i < 4; ++i) {
    const int k = (pb + (i << 2)) << 1;
    b[2 * i]     = *(const float4*)(base + (size_t)k * ldg + n4);
    b[2 * i + 1] = *(const float4*)(base + (size_t)(k + 1) * ldg + n4);
  }
}
// ---- convert + stage into LDS (A row-major, B transposed: Blds[n][k]) ------
__device__ __forceinline__ void stageA(const float4 a[4], unsigned short* Alds, int tid) {
  const int m = tid >> 3, kq = (tid & 7) << 2;
  #pragma unroll
  for (int i = 0; i < 4; ++i) {
    uint2 u;
    u.x = pack2bf(a[i].x, a[i].y);
    u.y = pack2bf(a[i].z, a[i].w);
    *(uint2*)(Alds + (m + (i << 5)) * LDT + kq) = u;
  }
}
__device__ __forceinline__ void stageB(const float4 b[8], unsigned short* Blds, int tid) {
  const int n4 = (tid & 63) << 2, pb = tid >> 6;
  #pragma unroll
  for (int i = 0; i < 4; ++i) {
    const int k2 = (pb + (i << 2)) << 1;            // even k of the pair
    const float4 fa = b[2 * i], fb = b[2 * i + 1];
    *(unsigned*)(Blds + (n4 + 0) * LDT + k2) = pack2bf(fa.x, fb.x);
    *(unsigned*)(Blds + (n4 + 1) * LDT + k2) = pack2bf(fa.y, fb.y);
    *(unsigned*)(Blds + (n4 + 2) * LDT + k2) = pack2bf(fa.z, fb.z);
    *(unsigned*)(Blds + (n4 + 3) * LDT + k2) = pack2bf(fa.w, fb.w);
  }
}

__device__ __forceinline__ v16bf frag16(const unsigned short* p0,
                                        const unsigned short* p1) {
  v8bf lo = *(const v8bf*)p0;
  v8bf hi = *(const v8bf*)p1;
  v16bf r;
  #pragma unroll
  for (int i = 0; i < 8; ++i) { r[i] = lo[i]; r[i + 8] = hi[i]; }
  return r;
}

// ---------------- small helper kernels ----------------
__global__ void zero4_kernel(float4* p) {
  size_t i = (size_t)blockIdx.x * blockDim.x + threadIdx.x;
  p[i] = make_float4(0.f, 0.f, 0.f, 0.f);
}

__global__ void degree_kernel(const int* __restrict__ ei, const int* __restrict__ et,
                              unsigned* cnt) {
  long e = (long)blockIdx.x * blockDim.x + threadIdx.x;
  if (e >= E_) return;
  atomicAdd(&cnt[(size_t)et[e] * N_ + ei[E_ + e]], 1u);
}

__global__ void normk_kernel(const int* __restrict__ ei, const int* __restrict__ et,
                             const unsigned* __restrict__ cnt, float* nrm) {
  long e = (long)blockIdx.x * blockDim.x + threadIdx.x;
  if (e >= E_) return;
  unsigned c = cnt[(size_t)et[e] * N_ + ei[E_ + e]];
  nrm[e] = 1.0f / (float)(c > 1u ? c : 1u);
}

// one wave per edge: gather h[src] (8 coalesced f32/lane), scatter 4 weighted
// copies into A[dst, b, :] with global f32 atomics.
__global__ __launch_bounds__(256) void scatter_kernel(
    const float* __restrict__ hin, const int* __restrict__ ei,
    const int* __restrict__ et, const float* __restrict__ nrm,
    const float* __restrict__ comp_l, float* Agg) {
  const int lane = threadIdx.x & 31;
  const long e = (long)blockIdx.x * 8 + (threadIdx.x >> 5);
  if (e >= E_) return;
  const int src = ei[e];
  const int dst = ei[E_ + e];
  const int r   = et[e];
  const float nm = nrm[e];
  const float c0 = nm * comp_l[r * 4 + 0];
  const float c1 = nm * comp_l[r * 4 + 1];
  const float c2 = nm * comp_l[r * 4 + 2];
  const float c3 = nm * comp_l[r * 4 + 3];
  const float* hs = hin + (size_t)src * H_;
  float* ab = Agg + (size_t)dst * (B_ * H_);
  #pragma unroll
  for (int j = 0; j < 8; ++j) {
    const int c = lane + j * 32;
    const float v = hs[c];
    atomicAdd(ab +   0 + c, c0 * v);
    atomicAdd(ab + 256 + c, c1 * v);
    atomicAdd(ab + 512 + c, c2 * v);
    atomicAdd(ab + 768 + c, c3 * v);
  }
}

// ---------------- projection GEMM: h = x_t @ Wp_t (per type) ----------------
// block = 256 thr (8 waves); block tile 128x256; wave tile 16x256 (16 WMMAs/kstep)
__global__ __launch_bounds__(256) void proj_kernel(const float* __restrict__ x,
                                                   const float* __restrict__ Wp,
                                                   float* __restrict__ h) {
  __shared__ __align__(16) unsigned short Alds[2][128 * LDT];
  __shared__ __align__(16) unsigned short Blds[2][256 * LDT];
  const int tid = threadIdx.x, lane = tid & 31, wr = tid >> 5;
  const int t = blockIdx.x >> 4;
  const int m0 = (blockIdx.x & 15) * 128;
  const float* Abase = x + (size_t)(t * NPER + m0) * D_;
  const float* Bbase = Wp + (size_t)t * D_ * H_;

  v8f acc[16];
  #pragma unroll
  for (int ct = 0; ct < 16; ++ct)
    #pragma unroll
    for (int i = 0; i < 8; ++i) acc[ct][i] = 0.f;

  const int aRow  = wr * 16 + (lane & 15);
  const int akoff = (lane >> 4) * 16;
  const int bkoff = (lane >> 4) * 16;
  constexpr int NK = D_ / 32;

  float4 ra[4], rb[8];
  glob_loadA(Abase, D_, tid, ra);
  glob_loadB(Bbase, H_, tid, rb);

  int cur = 0;
  for (int ks = 0; ks < NK; ++ks) {
    stageA(ra, Alds[cur], tid);
    stageB(rb, Blds[cur], tid);
    __syncthreads();
    if (ks + 1 < NK) {  // issue next-step global loads; they overlap the WMMAs
      glob_loadA(Abase + (ks + 1) * 32, D_, tid, ra);
      glob_loadB(Bbase + (size_t)(ks + 1) * 32 * H_, H_, tid, rb);
    }
    const unsigned short* Ac = Alds[cur];
    const unsigned short* Bc = Blds[cur];
    v16bf af = frag16(&Ac[aRow * LDT + akoff], &Ac[aRow * LDT + akoff + 8]);
    const unsigned short* bp0 = &Bc[(lane & 15) * LDT + bkoff];
    v16bf bcur = frag16(bp0, bp0 + 8);
    #pragma unroll
    for (int ct = 0; ct < 16; ++ct) {
      v16bf bnext = bcur;
      if (ct < 15) {  // software-pipeline: load frag ct+1 under WMMA ct
        const unsigned short* bp = &Bc[((ct + 1) * 16 + (lane & 15)) * LDT + bkoff];
        bnext = frag16(bp, bp + 8);
      }
      acc[ct] = __builtin_amdgcn_wmma_f32_16x16x32_bf16(
          false, af, false, bcur, (short)0, acc[ct], false, false);
      bcur = bnext;
    }
    cur ^= 1;
  }

  const int rowb = t * NPER + m0 + wr * 16 + (lane >> 4) * 8;
  const int colb = lane & 15;
  #pragma unroll
  for (int ct = 0; ct < 16; ++ct)
    #pragma unroll
    for (int r2 = 0; r2 < 8; ++r2)
      h[(size_t)(rowb + r2) * H_ + ct * 16 + colb] = acc[ct][r2];
}

// --------- layer GEMM: out = [h|A] @ [root;basis] + bias, then LN+ReLU ------
__device__ __forceinline__ void layer_srcs(int k0, const float* hin, const float* Agg,
                                           const float* root_l, const float* basis_l,
                                           int m0, const float*& Ab, int& lda,
                                           const float*& Bb) {
  if (k0 < 256) {
    Ab = hin + (size_t)m0 * H_ + k0;                lda = H_;
    Bb = root_l + (size_t)k0 * H_;
  } else {
    Ab = Agg + (size_t)m0 * (B_ * H_) + (k0 - 256); lda = B_ * H_;
    Bb = basis_l + (size_t)(k0 - 256) * H_;
  }
}

__global__ __launch_bounds__(256) void layer_fused_kernel(
    const float* __restrict__ hin, const float* __restrict__ Agg,
    const float* __restrict__ root_l, const float* __restrict__ basis_l,
    const float* __restrict__ bias_l, const float* __restrict__ gamma_l,
    const float* __restrict__ beta_l, float* __restrict__ hout) {
  __shared__ __align__(16) unsigned short Alds[2][128 * LDT];
  __shared__ __align__(16) unsigned short Blds[2][256 * LDT];
  const int tid = threadIdx.x, lane = tid & 31, wr = tid >> 5;
  const int m0 = blockIdx.x * 128;

  v8f acc[16];
  #pragma unroll
  for (int ct = 0; ct < 16; ++ct)
    #pragma unroll
    for (int i = 0; i < 8; ++i) acc[ct][i] = 0.f;

  const int aRow  = wr * 16 + (lane & 15);
  const int akoff = (lane >> 4) * 16;
  const int bkoff = (lane >> 4) * 16;
  constexpr int NK = 1280 / 32;  // K = H + B*H; 32-tiles never straddle sources

  const float* Ab; int lda; const float* Bb;
  layer_srcs(0, hin, Agg, root_l, basis_l, m0, Ab, lda, Bb);
  float4 ra[4], rb[8];
  glob_loadA(Ab, lda, tid, ra);
  glob_loadB(Bb, H_, tid, rb);

  int cur = 0;
  for (int ks = 0; ks < NK; ++ks) {
    stageA(ra, Alds[cur], tid);
    stageB(rb, Blds[cur], tid);
    __syncthreads();
    if (ks + 1 < NK) {
      layer_srcs((ks + 1) * 32, hin, Agg, root_l, basis_l, m0, Ab, lda, Bb);
      glob_loadA(Ab, lda, tid, ra);
      glob_loadB(Bb, H_, tid, rb);
    }
    const unsigned short* Ac = Alds[cur];
    const unsigned short* Bc = Blds[cur];
    v16bf af = frag16(&Ac[aRow * LDT + akoff], &Ac[aRow * LDT + akoff + 8]);
    const unsigned short* bp0 = &Bc[(lane & 15) * LDT + bkoff];
    v16bf bcur = frag16(bp0, bp0 + 8);
    #pragma unroll
    for (int ct = 0; ct < 16; ++ct) {
      v16bf bnext = bcur;
      if (ct < 15) {
        const unsigned short* bp = &Bc[((ct + 1) * 16 + (lane & 15)) * LDT + bkoff];
        bnext = frag16(bp, bp + 8);
      }
      acc[ct] = __builtin_amdgcn_wmma_f32_16x16x32_bf16(
          false, af, false, bcur, (short)0, acc[ct], false, false);
      bcur = bnext;
    }
    cur ^= 1;
  }

  // Epilogue: +bias, LayerNorm over H=256 (16 in-lane + shfl over 16-lane half),
  // then ReLU. Wave owns rows [wr*16, wr*16+16) completely.
  const int colb = lane & 15;
  float s[8], sq[8];
  #pragma unroll
  for (int r2 = 0; r2 < 8; ++r2) { s[r2] = 0.f; sq[r2] = 0.f; }
  #pragma unroll
  for (int ct = 0; ct < 16; ++ct) {
    const float bv = bias_l[ct * 16 + colb];
    #pragma unroll
    for (int r2 = 0; r2 < 8; ++r2) {
      const float v = acc[ct][r2] + bv;
      acc[ct][r2] = v; s[r2] += v; sq[r2] += v * v;
    }
  }
  #pragma unroll
  for (int r2 = 0; r2 < 8; ++r2) {
    #pragma unroll
    for (int off = 1; off < 16; off <<= 1) {
      s[r2]  += __shfl_xor(s[r2],  off, 32);
      sq[r2] += __shfl_xor(sq[r2], off, 32);
    }
    const float mean = s[r2] * (1.f / 256.f);
    const float var  = sq[r2] * (1.f / 256.f) - mean * mean;
    s[r2]  = mean;
    sq[r2] = rsqrtf(var + 1e-5f);
  }
  const int rowb = m0 + wr * 16 + (lane >> 4) * 8;
  #pragma unroll
  for (int ct = 0; ct < 16; ++ct) {
    const int col = ct * 16 + colb;
    const float g = gamma_l[col], be = beta_l[col];
    #pragma unroll
    for (int r2 = 0; r2 < 8; ++r2) {
      const float v = g * (acc[ct][r2] - s[r2]) * sq[r2] + be;
      hout[(size_t)(rowb + r2) * H_ + col] = fmaxf(v, 0.f);
    }
  }
}

// ---------------------------------------------------------------------------
extern "C" void kernel_launch(void* const* d_in, const int* in_sizes, int n_in,
                              void* d_out, int out_size, void* d_ws, size_t ws_size,
                              hipStream_t stream) {
  const float* x     = (const float*)d_in[0];
  const float* Wp    = (const float*)d_in[1];
  const float* basis = (const float*)d_in[2];
  const float* comp  = (const float*)d_in[3];
  const float* root  = (const float*)d_in[4];
  const float* bias  = (const float*)d_in[5];
  const float* gamma = (const float*)d_in[6];
  const float* beta  = (const float*)d_in[7];
  const int*   ei    = (const int*)d_in[8];
  const int*   et    = (const int*)d_in[9];
  float* out = (float*)d_out;

  // workspace carve-up (~442 MB)
  char* ws = (char*)d_ws;
  float*    h0  = (float*)ws;    ws += (size_t)N_ * H_ * 4;
  float*    h1  = (float*)ws;    ws += (size_t)N_ * H_ * 4;
  float*    Agg = (float*)ws;    ws += (size_t)N_ * B_ * H_ * 4;
  unsigned* cnt = (unsigned*)ws; ws += (size_t)R_ * N_ * 4;
  float*    nrm = (float*)ws;    ws += (size_t)E_ * 4;

  // degree counts + per-edge mean-normalization
  zero4_kernel<<<((size_t)R_ * N_) / 1024, 256, 0, stream>>>((float4*)cnt);
  degree_kernel<<<(int)(E_ / 256), 256, 0, stream>>>(ei, et, cnt);
  normk_kernel<<<(int)(E_ / 256), 256, 0, stream>>>(ei, et, cnt, nrm);

  // per-type input projection
  proj_kernel<<<T_ * (NPER / 128), 256, 0, stream>>>(x, Wp, h0);

  const float* hin = h0;
  for (int l = 0; l < 2; ++l) {
    float* hout = (l == 0) ? h1 : out;
    zero4_kernel<<<((size_t)N_ * B_ * H_) / 1024, 256, 0, stream>>>((float4*)Agg);
    scatter_kernel<<<(int)(E_ / 8), 256, 0, stream>>>(
        hin, ei, et, nrm, comp + (size_t)l * R_ * B_, Agg);
    layer_fused_kernel<<<N_ / 128, 256, 0, stream>>>(
        hin, Agg, root + (size_t)l * H_ * H_, basis + (size_t)l * B_ * H_ * H_,
        bias + l * H_, gamma + l * H_, beta + l * H_, hout);
    hin = hout;
  }
}
